// SetConv1dEncoder_75943611728141
// MI455X (gfx1250) — compile-verified, matrix operand order checked by reading, not run on hardware
//
#include <hip/hip_runtime.h>
#include <hip/hip_bf16.h>

typedef __attribute__((ext_vector_type(16))) _Float16 v16h;
typedef __attribute__((ext_vector_type(8)))  _Float16 v8h;
typedef __attribute__((ext_vector_type(8)))  float    v8f;

// Problem constants (from reference setup): B=4, N=2048, C=8, G=8192
constexpr int kN = 2048;
constexpr int kG = 8192;
constexpr int kZRows = 10;          // density + 8 data channels + 1 zero pad row
constexpr int kZStride = kN + 8;    // halfs; 4112B row stride -> conflict-free banks
constexpr float kLog2e = 1.4426950408889634f;

__global__ __launch_bounds__(256)
void SetConv1dEncoder_75943611728141_kernel(const float* __restrict__ xz,
                                            const float* __restrict__ z,
                                            const float* __restrict__ x_grid,
                                            const float* __restrict__ log_scale,
                                            float* __restrict__ out) {
    __shared__ _Float16 sZ[kZRows * kZStride];      // z_aug^T, f16
    __shared__ __align__(16) float2 sPQ[kN];        // per-n: {p = c*x^2, q = -2c*x}

    const int tid    = threadIdx.x;
    const int b      = blockIdx.x >> 6;             // 64 blocks per batch
    const int gBlock = (blockIdx.x & 63) * 128;     // 8 waves x 16 rows

    // c folds log2(e): weight = exp2(c*(xg-x)^2), c = -0.5*log2e/s^2  (c < 0)
    const float ls    = log_scale[0];
    const float coef2 = -0.5f * __expf(-2.0f * ls) * kLog2e;

    // ---- Stage z_aug^T and the quadratic-expansion terms of xz into LDS ----
    for (int idx = tid; idx < kZRows * kN; idx += 256) {
        const int c = idx >> 11;                    // 0..9
        const int n = idx & (kN - 1);
        float v = 1.0f;                             // c==0: density channel of ones
        if (c >= 1 && c <= 8) v = z[(b * kN + n) * 8 + (c - 1)];
        if (c == 9)           v = 0.0f;             // zero pad (cols 9..15 clamp here)
        sZ[c * kZStride + n] = (_Float16)v;
    }
    for (int n = tid; n < kN; n += 256) {
        const float x = xz[b * kN + n];
        sPQ[n] = make_float2(coef2 * x * x, -2.0f * coef2 * x);
    }
    __syncthreads();

    const int wave  = tid >> 5;
    const int lane  = tid & 31;
    const int col   = lane & 15;     // A row index / B column / D column
    const int hi    = lane >> 4;     // lane half selects K sub-block
    const int kOff  = hi * 8;
    const int gTile = gBlock + wave * 16;

    const float xg = x_grid[gTile + col];           // grid point for this A row
    const float r0 = coef2 * xg * xg;

    // Copy x_grid into the first G outputs (tuple element 0); batch-0 blocks cover all g.
    if (b == 0 && lane < 16) out[gTile + lane] = x_grid[gTile + lane];

    const int cB = (col <= 9) ? col : 9;            // pad columns read the zero row

    v8f acc = {};
    for (int nb = 0; nb < kN; nb += 32) {
        // ---- A fragment: 16 RBF weights; exponent = fma(q,xg,p) + r0 ----
        float pqv[32];
        const float4* pp0 = (const float4*)&sPQ[nb + kOff];       // elems 0..7
        const float4* pp1 = (const float4*)&sPQ[nb + 16 + kOff];  // elems 8..15
        *(float4*)&pqv[0]  = pp0[0];
        *(float4*)&pqv[4]  = pp0[1];
        *(float4*)&pqv[8]  = pp0[2];
        *(float4*)&pqv[12] = pp0[3];
        *(float4*)&pqv[16] = pp1[0];
        *(float4*)&pqv[20] = pp1[1];
        *(float4*)&pqv[24] = pp1[2];
        *(float4*)&pqv[28] = pp1[3];
        v16h a;
#pragma unroll
        for (int i = 0; i < 16; ++i) {
            const float e = __builtin_fmaf(pqv[2 * i + 1], xg, pqv[2 * i]) + r0;
            a[i] = (_Float16)__builtin_amdgcn_exp2f(e);   // raw v_exp_f32
        }
        // ---- B fragment: two 16-byte LDS loads of z_aug^T ----
        const v8h b0 = *(const v8h*)&sZ[cB * kZStride + nb + kOff];
        const v8h b1 = *(const v8h*)&sZ[cB * kZStride + nb + 16 + kOff];
        v16h bf;
#pragma unroll
        for (int i = 0; i < 8; ++i) { bf[i] = b0[i]; bf[8 + i] = b1[i]; }

        acc = __builtin_amdgcn_wmma_f32_16x16x32_f16(
            /*neg_a=*/false, a, /*neg_b=*/false, bf,
            /*c_mod=*/(short)0, acc, /*reuse_a=*/false, /*reuse_b=*/false);
    }

    // ---- Epilogue: divide data channels by density, store channels-first ----
    float* outD = out + kG;                         // tuple element 1: [B, 9, G]
    const int rowHi = hi * 8;
#pragma unroll
    for (int r = 0; r < 8; ++r) {
        const float v    = acc[r];
        const float dens = __shfl(v, lane & 16, 32);   // column-0 value for this row
        const int   g    = gTile + rowHi + r;
        if (col <= 8) {
            // channel 0 stores raw density; channels 1..8 scale by fast reciprocal
            const float res = (col == 0)
                                  ? v
                                  : v * __builtin_amdgcn_rcpf(dens + 1e-8f);
            outD[(b * 9 + col) * kG + g] = res;
        }
    }
}

extern "C" void kernel_launch(void* const* d_in, const int* in_sizes, int n_in,
                              void* d_out, int out_size, void* d_ws, size_t ws_size,
                              hipStream_t stream) {
    const float* xz        = (const float*)d_in[0];
    const float* z         = (const float*)d_in[1];
    const float* x_grid    = (const float*)d_in[2];
    const float* log_scale = (const float*)d_in[3];
    float* out             = (float*)d_out;
    // 4 batches x 64 tiles of 128 grid rows; 256 threads = 8 wave32
    SetConv1dEncoder_75943611728141_kernel<<<256, 256, 0, stream>>>(
        xz, z, x_grid, log_scale, out);
}
